// selective_matching_ver_20280835572217
// MI455X (gfx1250) — compile-verified, matrix operand order checked by reading, not run on hardware
//
#include <hip/hip_runtime.h>
#include <hip/hip_bf16.h>

// ---------------- fixed problem config ----------------
#define NN    2
#define AN    5
#define CCH   64
#define HH    96
#define WW    96
#define KNBR  6
#define PSH   8
#define PNH   12          // H / PSH
#define BB    (NN*PNH)    // 24
#define CCDIM (CCH*AN*PSH) // 2560
#define PP    (AN*WW)     // 480
#define IMG   (NN*AN*WW)  // 960
#define HW    (HH*WW)     // 9216

typedef __attribute__((ext_vector_type(2)))  float   v2f;
typedef __attribute__((ext_vector_type(8)))  float   v8f;
typedef __attribute__((ext_vector_type(16))) __bf16  v16bf;

union VU16 { v16bf v; unsigned int u[8]; };

__device__ __forceinline__ float bf16_to_f32(unsigned short h) {
    unsigned int u = ((unsigned int)h) << 16;
    return __uint_as_float(u);
}
__device__ __forceinline__ unsigned short f32_to_bf16(float f) {
    unsigned int u = __float_as_uint(f);
    unsigned int r = (u + 0x7FFFu + ((u >> 16) & 1u)) >> 16;   // RNE
    return (unsigned short)r;
}
__device__ __forceinline__ float lrelu(float x) { return x > 0.0f ? x : 0.1f * x; }

// ---------------------------------------------------------------------------
// K1: pack lf_fea into K-contiguous bf16 lfT[b][p][cc]
//   lfT[(n,pnh)][(v,w)][(c,u,psh)] = lf_fea[(n*25+u*5+v), c, pnh*8+psh, w]
// ---------------------------------------------------------------------------
__global__ __launch_bounds__(256) void pack_lfT(const float* __restrict__ lf_fea,
                                                unsigned short* __restrict__ lfT) {
    long i = (long)blockIdx.x * 256 + threadIdx.x;
    const long total = (long)BB * PP * CCDIM;
    if (i >= total) return;
    int cc = (int)(i % CCDIM); long t = i / CCDIM;
    int p  = (int)(t % PP);    int b = (int)(t / PP);
    int n = b / PNH, pnh = b % PNH;
    int v = p / WW,  w   = p % WW;
    int c = cc / (AN*PSH); int r = cc % (AN*PSH);
    int u = r / PSH;       int psh = r % PSH;
    float val = lf_fea[(size_t)((n*25 + u*5 + v)*CCH + c) * HW + (pnh*PSH + psh)*WW + w];
    lfT[i] = f32_to_bf16(val);
}

// ---------------------------------------------------------------------------
// K2: sq[b][p] = sum_cc lfT^2   (one wave32 per row)
// ---------------------------------------------------------------------------
__global__ __launch_bounds__(256) void sqnorm(const unsigned short* __restrict__ lfT,
                                              float* __restrict__ sq) {
    int wid  = (int)((blockIdx.x * 256 + threadIdx.x) >> 5);
    int lane = threadIdx.x & 31;
    if (wid >= BB * PP) return;
    const unsigned short* row = lfT + (size_t)wid * CCDIM;
    float s = 0.0f;
    for (int i = lane; i < CCDIM; i += 32) {
        float f = bf16_to_f32(row[i]);
        s += f * f;
    }
    for (int off = 16; off > 0; off >>= 1) s += __shfl_down(s, off, 32);
    if (lane == 0) sq[wid] = s;
}

// ---------------------------------------------------------------------------
// K3: Gram + distance.  One wave per 32x32 macro-tile (2x2 WMMA tiles) so
// each K-step loads 2 A + 2 B fragments and issues 4 WMMAs (2x register
// reuse vs 1 tile/wave).
//   dist[b][p][q] = sq[p] + sq[q] - 2 * <lf[:,p], lf[:,q]>
// bf16 WMMA 16x16x32, K swept over 2560.
// 16-bit A 16x32 lane layout: lane<16 -> M=lane, K in {0..7,16..23};
//                             lane>=16 -> M=lane-16, K in {8..15,24..31}.
// lfT rows are K-contiguous, so each bf16 pair is one dword load.
// ---------------------------------------------------------------------------
__global__ __launch_bounds__(32) void gram_dist(const unsigned short* __restrict__ lfT,
                                                const float* __restrict__ sq,
                                                float* __restrict__ dist) {
    int b  = blockIdx.z;
    int p0 = blockIdx.x * 32;
    int q0 = blockIdx.y * 32;
    int lane = threadIdx.x;
    int m  = lane & 15;
    int hi = lane >> 4;              // 0 or 1

    const unsigned short* rowA0 = lfT + (size_t)(b*PP + p0 + m) * CCDIM;
    const unsigned short* rowA1 = rowA0 + (size_t)16 * CCDIM;
    const unsigned short* rowB0 = lfT + (size_t)(b*PP + q0 + m) * CCDIM;
    const unsigned short* rowB1 = rowB0 + (size_t)16 * CCDIM;

    v8f acc00 = {}, acc01 = {}, acc10 = {}, acc11 = {};
    for (int cc0 = 0; cc0 < CCDIM; cc0 += 32) {
        int k0 = cc0 + hi * 8;
        VU16 a0, a1, b0, b1;
#pragma unroll
        for (int j = 0; j < 8; ++j) {
            int off = k0 + ((j < 4) ? (2*j) : (8 + 2*j));   // K pairs {0..7} then {16..23} (+8 for hi lanes)
            a0.u[j] = *(const unsigned int*)(rowA0 + off);
            a1.u[j] = *(const unsigned int*)(rowA1 + off);
            b0.u[j] = *(const unsigned int*)(rowB0 + off);
            b1.u[j] = *(const unsigned int*)(rowB1 + off);
        }
        acc00 = __builtin_amdgcn_wmma_f32_16x16x32_bf16(false, a0.v, false, b0.v,
                                                        (short)0, acc00, false, false);
        acc01 = __builtin_amdgcn_wmma_f32_16x16x32_bf16(false, a0.v, false, b1.v,
                                                        (short)0, acc01, false, false);
        acc10 = __builtin_amdgcn_wmma_f32_16x16x32_bf16(false, a1.v, false, b0.v,
                                                        (short)0, acc10, false, false);
        acc11 = __builtin_amdgcn_wmma_f32_16x16x32_bf16(false, a1.v, false, b1.v,
                                                        (short)0, acc11, false, false);
    }

    float sqq0 = sq[b*PP + q0 + m];
    float sqq1 = sq[b*PP + q0 + 16 + m];
#pragma unroll
    for (int r = 0; r < 8; ++r) {
        int M = r + 8*hi;
        float sqp0 = sq[b*PP + p0 + M];
        float sqp1 = sq[b*PP + p0 + 16 + M];
        dist[(size_t)(b*PP + p0 + M)      * PP + q0 + m]      = sqp0 + sqq0 - 2.0f * acc00[r];
        dist[(size_t)(b*PP + p0 + M)      * PP + q0 + 16 + m] = sqp0 + sqq1 - 2.0f * acc01[r];
        dist[(size_t)(b*PP + p0 + 16 + M) * PP + q0 + m]      = sqp1 + sqq0 - 2.0f * acc10[r];
        dist[(size_t)(b*PP + p0 + 16 + M) * PP + q0 + 16 + m] = sqp1 + sqq1 - 2.0f * acc11[r];
    }
}

// ---------------------------------------------------------------------------
// K4: stable top-6 smallest (ascending, lower index wins ties) per row
// ---------------------------------------------------------------------------
__global__ __launch_bounds__(256) void topk6(const float* __restrict__ dist,
                                             int* __restrict__ idx) {
    int t = blockIdx.x * 256 + threadIdx.x;
    if (t >= BB * PP) return;
    const float* row = dist + (size_t)t * PP;
    float bd[KNBR]; int bi[KNBR];
#pragma unroll
    for (int j = 0; j < KNBR; ++j) { bd[j] = 3.4e38f; bi[j] = 0; }
    for (int q = 0; q < PP; ++q) {
        float d = row[q];
        if (d < bd[KNBR-1]) {
            int j = KNBR - 1;
            while (j > 0 && d < bd[j-1]) { bd[j] = bd[j-1]; bi[j] = bi[j-1]; --j; }
            bd[j] = d; bi[j] = q;
        }
    }
#pragma unroll
    for (int j = 0; j < KNBR; ++j) idx[t*KNBR + j] = bi[j];
}

// ---------------------------------------------------------------------------
// K5: agg1 = leaky_relu(1x1 conv over gathered neighbors)
// GEMM: D[co, col] = sum_{kk<384} w1[co,kk] * gathered[kk, col]
// col enumerates (n,v,w,u,pnh,psh); gather index = idx[(n,pnh)][(v,w)][k]
// f32 WMMA 16x16x4: lane<16 -> K {0,1}; lane>=16 -> K {2,3}.
// w1 (64x384 f32 = 96KB) staged in dynamic LDS.
// ---------------------------------------------------------------------------
__global__ __launch_bounds__(256) void agg1_gemm(const float* __restrict__ lf_fea,
                                                 const float* __restrict__ w1,
                                                 const int*   __restrict__ idxbuf,
                                                 float*       __restrict__ a1) {
    extern __shared__ float lds[];                   // 64*384 floats
    for (int i = threadIdx.x; i < CCH * (CCH*KNBR); i += 256) lds[i] = w1[i];
    __syncthreads();

    int wave = threadIdx.x >> 5, lane = threadIdx.x & 31;
    int co0    = (wave & 3) * 16;
    int colgrp = wave >> 2;

    long col = (long)blockIdx.x * 32 + colgrp * 16 + (lane & 15);
    int psh = (int)(col & 7); long t = col >> 3;
    int pnh = (int)(t % PNH); t /= PNH;
    int u   = (int)(t % AN);  t /= AN;
    int w   = (int)(t % WW);  t /= WW;
    int v   = (int)(t % AN);
    int n   = (int)(t / AN);

    int bidx = n*PNH + pnh;
    int p    = v*WW + w;
    long base_k[KNBR];
#pragma unroll
    for (int k = 0; k < KNBR; ++k) {
        int q  = idxbuf[(bidx*PP + p)*KNBR + k];
        int vq = q / WW, wq = q - vq*WW;
        base_k[k] = (long)((n*25 + u*5 + vq) * CCH) * HW + (pnh*PSH + psh)*WW + wq;
    }

    int e0   = (lane < 16) ? 0 : 2;
    int arow = (co0 + (lane & 15)) * (CCH*KNBR);
    v8f acc = {};
    for (int it = 0; it < (CCH*KNBR)/4; ++it) {
        int kk0 = it * 4;
        int k   = kk0 >> 6;
        int c0  = (kk0 & 63) + e0;
        v2f A  = { lds[arow + kk0 + e0], lds[arow + kk0 + e0 + 1] };
        v2f Bv = { lf_fea[base_k[k] + (long)c0 * HW],
                   lf_fea[base_k[k] + (long)(c0 + 1) * HW] };
        acc = __builtin_amdgcn_wmma_f32_16x16x4_f32(false, A, false, Bv,
                                                    (short)0, acc, false, false);
    }

    int h = pnh*PSH + psh;
    long nvw = (long)((n*AN + v)*WW + w);
#pragma unroll
    for (int r = 0; r < 8; ++r) {
        int co = co0 + r + 8*(lane >> 4);
        a1[((nvw*CCH + co)*AN + u)*HH + h] = lrelu(acc[r]);
    }
}

// ---------------------------------------------------------------------------
// K6: agg2 = leaky_relu(3x3 conv, pad 1) over epi = concat(lf_ver, a1),
// fused final transpose into d_out.
// GEMM K = 128*9 = 1152; w2 (64x1152 f32 = 288KB) staged in dynamic LDS
// (CDNA5 allows up to 320KB LDS per workgroup).
// ---------------------------------------------------------------------------
__global__ __launch_bounds__(256) void agg2_conv(const float* __restrict__ lf_fea,
                                                 const float* __restrict__ w2,
                                                 const float* __restrict__ a1,
                                                 float*       __restrict__ out) {
    extern __shared__ float lds[];                   // 64*1152 floats
    for (int i = threadIdx.x; i < CCH * (2*CCH*9); i += 256) lds[i] = w2[i];
    __syncthreads();

    int wave = threadIdx.x >> 5, lane = threadIdx.x & 31;
    int co0    = (wave & 3) * 16;
    int colgrp = wave >> 2;

    int img = blockIdx.x / 15;                        // 0..959  == (n*5+v)*96+w
    int blk = blockIdx.x % 15;
    int col = blk*32 + colgrp*16 + (lane & 15);       // 0..479 == u*96+h
    int u = col / HH, h = col % HH;
    int w = img % WW; int t = img / WW;
    int v = t % AN;   int n = t / AN;

    int e0   = (lane < 16) ? 0 : 2;
    int arow = (co0 + (lane & 15)) * (2*CCH*9);
    v8f acc = {};
    for (int it = 0; it < (2*CCH*9)/4; ++it) {
        int kk0 = it * 4;
        float bv[2];
#pragma unroll
        for (int e = 0; e < 2; ++e) {
            int kk = kk0 + e0 + e;
            int ci = kk / 9; int r9 = kk - ci*9;
            int du = r9 / 3; int dh = r9 - du*3;
            int uu = u + du - 1, hh = h + dh - 1;
            float val = 0.0f;
            if (uu >= 0 && uu < AN && hh >= 0 && hh < HH) {
                if (ci < CCH)
                    val = lf_fea[(size_t)((n*25 + uu*5 + v)*CCH + ci) * HW + hh*WW + w];
                else
                    val = a1[(((size_t)img*CCH + (ci - CCH))*AN + uu)*HH + hh];
            }
            bv[e] = val;
        }
        v2f A  = { lds[arow + kk0 + e0], lds[arow + kk0 + e0 + 1] };
        v2f Bv = { bv[0], bv[1] };
        acc = __builtin_amdgcn_wmma_f32_16x16x4_f32(false, A, false, Bv,
                                                    (short)0, acc, false, false);
    }

#pragma unroll
    for (int r = 0; r < 8; ++r) {
        int co = co0 + r + 8*(lane >> 4);
        out[(size_t)((n*25 + u*5 + v)*CCH + co) * HW + h*WW + w] = lrelu(acc[r]);
    }
}

// ---------------------------------------------------------------------------
extern "C" void kernel_launch(void* const* d_in, const int* in_sizes, int n_in,
                              void* d_out, int out_size, void* d_ws, size_t ws_size,
                              hipStream_t stream) {
    (void)in_sizes; (void)n_in; (void)out_size; (void)ws_size;
    const float* lf_fea = (const float*)d_in[0];
    const float* w1     = (const float*)d_in[1];
    const float* w2     = (const float*)d_in[2];
    float* out          = (float*)d_out;

    // workspace layout (256B aligned)
    char* ws = (char*)d_ws;
    size_t off = 0;
    auto take = [&](size_t bytes) {
        void* p = ws + off;
        off += (bytes + 255) & ~(size_t)255;
        return p;
    };
    unsigned short* lfT = (unsigned short*)take((size_t)BB * PP * CCDIM * sizeof(unsigned short)); // 59 MB
    float*          sq  = (float*)take((size_t)BB * PP * sizeof(float));
    float*          dst = (float*)take((size_t)BB * PP * PP * sizeof(float));                      // 22 MB
    int*            idx = (int*)take((size_t)BB * PP * KNBR * sizeof(int));
    float*          a1  = (float*)take((size_t)IMG * CCH * AN * HH * sizeof(float));               // 118 MB

    {   // K1 pack
        long total = (long)BB * PP * CCDIM;
        int blocks = (int)((total + 255) / 256);
        pack_lfT<<<blocks, 256, 0, stream>>>(lf_fea, lfT);
    }
    {   // K2 squared norms: one wave per (b,p) row
        int waves = BB * PP;
        int blocks = (waves * 32 + 255) / 256;
        sqnorm<<<blocks, 256, 0, stream>>>(lfT, sq);
    }
    {   // K3 gram + dist: one wave per 32x32 macro-tile (2x2 WMMA tiles)
        dim3 grid(PP/32, PP/32, BB);
        gram_dist<<<grid, 32, 0, stream>>>(lfT, sq, dst);
    }
    {   // K4 top-6
        int blocks = (BB * PP + 255) / 256;
        topk6<<<blocks, 256, 0, stream>>>(dst, idx);
    }
    {   // K5 agg1 GEMM: 460800 columns / 32 per block
        int blocks = (IMG * AN * HH) / 32;   // 14400
        size_t shmem = (size_t)CCH * (CCH*KNBR) * sizeof(float);   // 96 KB
        agg1_gemm<<<blocks, 256, shmem, stream>>>(lf_fea, w1, idx, a1);
    }
    {   // K6 agg2 conv + output transpose
        int blocks = IMG * 15;               // 14400
        size_t shmem = (size_t)CCH * (2*CCH*9) * sizeof(float);    // 288 KB
        agg2_conv<<<blocks, 256, shmem, stream>>>(lf_fea, w2, a1, out);
    }
}
